// Attention3D_16578573763056
// MI455X (gfx1250) — matrix operand order repaired and checked by reading, not compiled, not run on hardware
//
#include <hip/hip_runtime.h>

typedef __attribute__((ext_vector_type(16))) __bf16 v16bf;
typedef __attribute__((ext_vector_type(8)))  float  v8f;

union V16 { uint4 u[2]; v16bf v; };

__device__ __forceinline__ unsigned short f2bf(float f) {
  unsigned u = __float_as_uint(f);
  u += 0x7FFFu + ((u >> 16) & 1u);          // round-to-nearest-even
  return (unsigned short)(u >> 16);
}

__device__ __forceinline__ v8f wmma_bf16(v16bf a, v16bf b, v8f c) {
  return __builtin_amdgcn_wmma_f32_16x16x32_bf16(false, a, false, b, (short)0, c,
                                                 false, false);
}

// CDNA5 async copy: global memory -> LDS, 16B per lane, tracked by ASYNCcnt.
__device__ __forceinline__ void async_ld_b128(unsigned lds_off, const void* gptr) {
  asm volatile("global_load_async_to_lds_b128 %0, %1, off"
               :: "v"(lds_off), "v"(gptr) : "memory");
}
__device__ __forceinline__ void wait_async0() {
  asm volatile("s_wait_asynccnt 0x0" ::: "memory");
}

// ---------------------------------------------------------------------------
// elementwise fp32 -> bf16
__global__ __launch_bounds__(256) void cvt_bf16(const float* __restrict__ s,
                                                unsigned short* __restrict__ d, int n) {
  int i = blockIdx.x * 256 + threadIdx.x;
  if (i < n) d[i] = f2bf(s[i]);
}

// src [K x N] fp32 row-major  ->  dst [N x K] bf16 (WMMA B-operand layout)
__global__ __launch_bounds__(256) void transpose_cvt(const float* __restrict__ s,
                                                     unsigned short* __restrict__ d,
                                                     int K, int N) {
  int i = blockIdx.x * 256 + threadIdx.x;
  if (i < K * N) {
    int k = i / N, n = i % N;
    d[(long)n * K + k] = f2bf(s[i]);
  }
}

// ---------------------------------------------------------------------------
// QKV GEMM: [16384 x 512] @ Bt[1536 x 512]^T -> scatter q/k/v [B,H,N,64] bf16
__global__ __launch_bounds__(256) void qkv_gemm(const unsigned short* __restrict__ A,
                                                const unsigned short* __restrict__ Bt,
                                                unsigned short* __restrict__ qo,
                                                unsigned short* __restrict__ ko,
                                                unsigned short* __restrict__ vo) {
  const int K = 512;
  int tid = threadIdx.x, wave = tid >> 5, lane = tid & 31;
  int l16 = lane & 15, hi = lane >> 4;
  int Rb = blockIdx.x * 128 + wave * 16;
  int Nb = blockIdx.y * 128;
  int lo = hi * 8;

  v8f acc[8];
#pragma unroll
  for (int t = 0; t < 8; t++)
#pragma unroll
    for (int e = 0; e < 8; e++) acc[t][e] = 0.f;

  const uint4* ap = (const uint4*)(A + (long)(Rb + l16) * K);
  for (int s = 0; s < K / 32; s++) {
    int k0 = 32 * s;
    V16 Av;
    Av.u[0] = ap[(k0 + lo) >> 3];
    Av.u[1] = ap[(k0 + lo + 16) >> 3];
#pragma unroll
    for (int t = 0; t < 8; t++) {
      const uint4* bp = (const uint4*)(Bt + (long)(Nb + t * 16 + l16) * K);
      int bb = k0 + hi * 16;
      V16 Bv;
      Bv.u[0] = bp[(k0 + lo) >> 3];
      Bv.u[1] = bp[(k0 + lo + 16) >> 3];
      acc[t] = wmma_bf16(Av.v, Bv.v, acc[t]);
    }
  }
#pragma unroll
  for (int t = 0; t < 8; t++) {
    int col = Nb + t * 16 + l16;
    int which = col >> 9, rem = col & 511;
    int h = rem >> 6, d = rem & 63;
    unsigned short* dst = (which == 0) ? qo : ((which == 1) ? ko : vo);
#pragma unroll
    for (int r = 0; r < 8; r++) {
      int row = Rb + r + hi * 8;
      int b = row >> 12, n = row & 4095;
      dst[(((long)(b * 8 + h)) * 4096 + n) * 64 + d] = f2bf(acc[t][r]);
    }
  }
}

// ---------------------------------------------------------------------------
// Proj GEMM: attn_o [16384 x 512] @ Bt[512 x 512]^T + bias -> out fp32
__global__ __launch_bounds__(256) void proj_gemm(const unsigned short* __restrict__ A,
                                                 const unsigned short* __restrict__ Bt,
                                                 const float* __restrict__ bias,
                                                 float* __restrict__ out) {
  const int K = 512;
  int tid = threadIdx.x, wave = tid >> 5, lane = tid & 31;
  int l16 = lane & 15, hi = lane >> 4;
  int Rb = blockIdx.x * 128 + wave * 16;
  int Nb = blockIdx.y * 128;
  int lo = hi * 8;

  v8f acc[8];
#pragma unroll
  for (int t = 0; t < 8; t++)
#pragma unroll
    for (int e = 0; e < 8; e++) acc[t][e] = 0.f;

  const uint4* ap = (const uint4*)(A + (long)(Rb + l16) * K);
  for (int s = 0; s < K / 32; s++) {
    int k0 = 32 * s;
    V16 Av;
    Av.u[0] = ap[(k0 + lo) >> 3];
    Av.u[1] = ap[(k0 + lo + 16) >> 3];
#pragma unroll
    for (int t = 0; t < 8; t++) {
      const uint4* bp = (const uint4*)(Bt + (long)(Nb + t * 16 + l16) * K);
      int bb = k0 + hi * 16;
      V16 Bv;
      Bv.u[0] = bp[(k0 + lo) >> 3];
      Bv.u[1] = bp[(k0 + lo + 16) >> 3];
      acc[t] = wmma_bf16(Av.v, Bv.v, acc[t]);
    }
  }
#pragma unroll
  for (int t = 0; t < 8; t++) {
    int col = Nb + t * 16 + l16;
    float bcol = bias[col];
#pragma unroll
    for (int r = 0; r < 8; r++) {
      int row = Rb + r + hi * 8;
      out[(long)row * 512 + col] = acc[t][r] + bcol;
    }
  }
}

// ---------------------------------------------------------------------------
// SR branch: depthwise 2x2/s2 conv + bias + LayerNorm(C) + linear -> x_ bf16 [B*M,64]
__global__ __launch_bounds__(256) void sr_branch(const float* __restrict__ x,
                                                 const float* __restrict__ wc,
                                                 const float* __restrict__ bc,
                                                 const float* __restrict__ g,
                                                 const float* __restrict__ be,
                                                 const float* __restrict__ wl,
                                                 const float* __restrict__ bl,
                                                 unsigned short* __restrict__ xo) {
  __shared__ float ybuf[512];
  __shared__ float red[16];
  __shared__ float part[4][64];
  int tid = threadIdx.x;
  int bm = blockIdx.x;
  int b = bm >> 10, m = bm & 1023;
  int i = m >> 5, j = m & 31;
  int n0 = (2 * i) * 64 + 2 * j;
  const float* xb = x + (long)b * 4096 * 512;

  float s = 0.f, sq = 0.f;
  for (int c = tid; c < 512; c += 256) {
    float y = bc[c] + xb[(long)n0 * 512 + c] * wc[c * 4 + 0] +
              xb[(long)(n0 + 1) * 512 + c] * wc[c * 4 + 1] +
              xb[(long)(n0 + 64) * 512 + c] * wc[c * 4 + 2] +
              xb[(long)(n0 + 65) * 512 + c] * wc[c * 4 + 3];
    ybuf[c] = y;
    s += y;
    sq += y * y;
  }
  for (int o = 16; o >= 1; o >>= 1) {
    s += __shfl_xor(s, o);
    sq += __shfl_xor(sq, o);
  }
  int wave = tid >> 5, lane = tid & 31;
  if (lane == 0) { red[wave] = s; red[8 + wave] = sq; }
  __syncthreads();
  float st = 0.f, sqt = 0.f;
  for (int w = 0; w < 8; w++) { st += red[w]; sqt += red[8 + w]; }
  float mu = st * (1.f / 512.f);
  float var = sqt * (1.f / 512.f) - mu * mu;
  float rs = rsqrtf(var + 1e-5f);
  for (int c = tid; c < 512; c += 256)
    ybuf[c] = (ybuf[c] - mu) * rs * g[c] + be[c];
  __syncthreads();
  int d = tid & 63, seg = tid >> 6;
  float acc = 0.f;
  for (int c = seg * 128; c < seg * 128 + 128; c++) acc += ybuf[c] * wl[c * 64 + d];
  part[seg][d] = acc;
  __syncthreads();
  if (tid < 64)
    xo[(long)bm * 64 + tid] =
        f2bf(bl[tid] + part[0][tid] + part[1][tid] + part[2][tid] + part[3][tid]);
}

// ---------------------------------------------------------------------------
// Generic flash attention (HD=64), bf16 in/out, fp32 online softmax.
// Block = 256 threads = 8 waves, each wave owns 16 query rows; keys chunked by 64.
#define LDSS 72
__global__ __launch_bounds__(256) void flash_attn(
    const unsigned short* __restrict__ qp, long q_bs, long q_hs, int q_rs,
    const unsigned short* __restrict__ kp, const unsigned short* __restrict__ vp,
    unsigned short* __restrict__ op, long o_bs, long o_hs, int o_rs,
    int nkeys, float scale, int heads) {
  __shared__ unsigned short Kc[64 * LDSS];
  __shared__ unsigned short Vt[64 * LDSS];
  __shared__ unsigned short Pl[8 * 16 * LDSS];

  int tid = threadIdx.x, wave = tid >> 5, lane = tid & 31;
  int l16 = lane & 15, hi = lane >> 4;
  int bh = blockIdx.y;
  int b = bh / heads, h = bh % heads;
  int lo = hi * 8;

  const unsigned short* qbase = qp + (long)b * q_bs + (long)h * q_hs;
  const unsigned short* kbase = kp + (long)bh * (long)nkeys * 64;
  const unsigned short* vbase = vp + (long)bh * (long)nkeys * 64;
  int qrow0 = blockIdx.x * 128 + wave * 16;

  // Q held in A-operand layout for both k-steps (dims 0..31 / 32..63)
  V16 A0, A1;
  const uint4* qrp = (const uint4*)(qbase + (long)(qrow0 + l16) * q_rs);
  A0.u[0] = qrp[lo >> 3];
  A0.u[1] = qrp[(lo + 16) >> 3];
  A1.u[0] = qrp[(32 + lo) >> 3];
  A1.u[1] = qrp[(48 + lo) >> 3];

  float mi[8], li[8];
  v8f oacc[4];
#pragma unroll
  for (int r = 0; r < 8; r++) { mi[r] = -1e30f; li[r] = 0.f; }
#pragma unroll
  for (int t = 0; t < 4; t++)
#pragma unroll
    for (int e = 0; e < 8; e++) oacc[t][e] = 0.f;

  unsigned short* myP = Pl + wave * 16 * LDSS;

  for (int c0 = 0; c0 < nkeys; c0 += 64) {
    __syncthreads();
    {  // stage K chunk natural [key][d] via async copy; V chunk transposed [d][key]
      const uint4* kg = (const uint4*)(kbase + (long)c0 * 64);
      const uint4* vg = (const uint4*)(vbase + (long)c0 * 64);
      for (int idx = tid; idx < 512; idx += 256) {
        int row = idx >> 3, c8 = idx & 7;
        async_ld_b128((unsigned)(unsigned long)&Kc[row * LDSS + c8 * 8], kg + idx);
        union { uint4 q; unsigned short s[8]; } vv;
        vv.q = vg[idx];
#pragma unroll
        for (int i2 = 0; i2 < 8; i2++) Vt[(c8 * 8 + i2) * LDSS + row] = vv.s[i2];
      }
      wait_async0();
    }
    __syncthreads();

    // S = Q @ Kchunk^T  (4 col-tiles x 2 k-steps)
    v8f sacc[4];
#pragma unroll
    for (int t = 0; t < 4; t++)
#pragma unroll
      for (int e = 0; e < 8; e++) sacc[t][e] = 0.f;
#pragma unroll
    for (int t = 0; t < 4; t++) {
      const uint4* br = (const uint4*)&Kc[(t * 16 + l16) * LDSS];
      int bb = hi * 16;
      V16 B0, B1;
      B0.u[0] = br[lo >> 3];
      B0.u[1] = br[(lo + 16) >> 3];
      B1.u[0] = br[(32 + lo) >> 3];
      B1.u[1] = br[(48 + lo) >> 3];
      sacc[t] = wmma_bf16(A0.v, B0.v, sacc[t]);
      sacc[t] = wmma_bf16(A1.v, B1.v, sacc[t]);
    }

    // online softmax (row stats per D-layout row; reduce within 16-lane half)
#pragma unroll
    for (int r = 0; r < 8; r++) {
      float s0 = sacc[0][r] * scale, s1 = sacc[1][r] * scale;
      float s2 = sacc[2][r] * scale, s3 = sacc[3][r] * scale;
      float vm = fmaxf(fmaxf(s0, s1), fmaxf(s2, s3));
      vm = fmaxf(vm, __shfl_xor(vm, 1));
      vm = fmaxf(vm, __shfl_xor(vm, 2));
      vm = fmaxf(vm, __shfl_xor(vm, 4));
      vm = fmaxf(vm, __shfl_xor(vm, 8));
      float newm = fmaxf(mi[r], vm);
      float alpha = __expf(mi[r] - newm);
      mi[r] = newm;
      float p0 = __expf(s0 - newm), p1 = __expf(s1 - newm);
      float p2 = __expf(s2 - newm), p3 = __expf(s3 - newm);
      float ps = p0 + p1 + p2 + p3;
      ps += __shfl_xor(ps, 1);
      ps += __shfl_xor(ps, 2);
      ps += __shfl_xor(ps, 4);
      ps += __shfl_xor(ps, 8);
      li[r] = li[r] * alpha + ps;
      int m = r + hi * 8;
      myP[m * LDSS + l16 + 0]  = f2bf(p0);
      myP[m * LDSS + l16 + 16] = f2bf(p1);
      myP[m * LDSS + l16 + 32] = f2bf(p2);
      myP[m * LDSS + l16 + 48] = f2bf(p3);
#pragma unroll
      for (int t = 0; t < 4; t++) oacc[t][r] *= alpha;
    }

    // O += P @ Vchunk  (P reloaded from per-wave LDS in A layout)
    V16 PA0, PA1;
    const uint4* pr = (const uint4*)&myP[l16 * LDSS];
    PA0.u[0] = pr[lo >> 3];
    PA0.u[1] = pr[(lo + 16) >> 3];
    PA1.u[0] = pr[(32 + lo) >> 3];
    PA1.u[1] = pr[(48 + lo) >> 3];
#pragma unroll
    for (int t = 0; t < 4; t++) {
      const uint4* vr = (const uint4*)&Vt[(t * 16 + l16) * LDSS];
      int bb = hi * 16;
      V16 B0, B1;
      B0.u[0] = vr[lo >> 3];
      B0.u[1] = vr[(lo + 16) >> 3];
      B1.u[0] = vr[(32 + lo) >> 3];
      B1.u[1] = vr[(48 + lo) >> 3];
      oacc[t] = wmma_bf16(PA0.v, B0.v, oacc[t]);
      oacc[t] = wmma_bf16(PA1.v, B1.v, oacc[t]);
    }
  }

  unsigned short* ob = op + (long)b * o_bs + (long)h * o_hs;
#pragma unroll
  for (int t = 0; t < 4; t++)
#pragma unroll
    for (int r = 0; r < 8; r++) {
      int m = qrow0 + r + hi * 8;
      int d = t * 16 + l16;
      ob[(long)m * o_rs + d] = f2bf(oacc[t][r] / li[r]);
    }
}

// ---------------------------------------------------------------------------
extern "C" void kernel_launch(void* const* d_in, const int* in_sizes, int n_in,
                              void* d_out, int out_size, void* d_ws, size_t ws_size,
                              hipStream_t stream) {
  (void)in_sizes; (void)n_in; (void)out_size; (void)ws_size;
  const float* x    = (const float*)d_in[0];
  const float* wqkv = (const float*)d_in[1];
  const float* wproj = (const float*)d_in[2];
  const float* bproj = (const float*)d_in[3];
  const float* wsrc = (const float*)d_in[4];
  const float* bsrc = (const float*)d_in[5];
  const float* nsc  = (const float*)d_in[6];
  const float* nbi  = (const float*)d_in[7];
  const float* wsrl = (const float*)d_in[8];
  const float* bsrl = (const float*)d_in[9];
  float* out = (float*)d_out;

  char* ws = (char*)d_ws;
  unsigned short* xb  = (unsigned short*)(ws + 0);           // 16384x512      16 MB
  unsigned short* wqt = (unsigned short*)(ws + 16777216);    // 1536x512       1.5 MB
  unsigned short* wpt = (unsigned short*)(ws + 18350080);    // 512x512        0.5 MB
  unsigned short* qb  = (unsigned short*)(ws + 18874368);    // [B,H,N,64]     16 MB
  unsigned short* kb  = (unsigned short*)(ws + 35651584);
  unsigned short* vb  = (unsigned short*)(ws + 52428800);
  unsigned short* xsr = (unsigned short*)(ws + 69206016);    // [B*M,64]       0.5 MB
  unsigned short* kcb = (unsigned short*)(ws + 69730304);    // [B,H,M,64]     4 MB
  unsigned short* vcb = (unsigned short*)(ws + 73924608);
  unsigned short* ao  = (unsigned short*)(ws + 78118912);    // [16384x512]    16 MB

  // 1) precision conversion / weight transposition
  cvt_bf16<<<(8388608 + 255) / 256, 256, 0, stream>>>(x, xb, 8388608);
  transpose_cvt<<<(786432 + 255) / 256, 256, 0, stream>>>(wqkv, wqt, 512, 1536);
  transpose_cvt<<<(262144 + 255) / 256, 256, 0, stream>>>(wproj, wpt, 512, 512);

  // 2) QKV projection GEMM -> q/k/v bf16 [B,H,N,64]
  qkv_gemm<<<dim3(128, 12), 256, 0, stream>>>(xb, wqt, qb, kb, vb);

  // 3) SR branch -> x_ bf16 [B*1024, 64]
  sr_branch<<<4096, 256, 0, stream>>>(x, wsrc, bsrc, nsc, nbi, wsrl, bsrl, xsr);

  // 4) softmax-weighted pooling (flash attn: q = x_, k = v = k or v), no scale
  flash_attn<<<dim3(8, 32), 256, 0, stream>>>(
      xsr, 65536L, 0L, 64, kb, kb, kcb, 524288L, 65536L, 64, 4096, 1.0f, 8);
  flash_attn<<<dim3(8, 32), 256, 0, stream>>>(
      xsr, 65536L, 0L, 64, vb, vb, vcb, 524288L, 65536L, 64, 4096, 1.0f, 8);

  // 5) main attention vs compressed K/V, scale = 1/sqrt(64); output interleaved [B,N,C]
  flash_attn<<<dim3(32, 32), 256, 0, stream>>>(
      qb, 2097152L, 262144L, 64, kcb, vcb, ao, 2097152L, 64L, 512, 1024, 0.125f, 8);

  // 6) output projection + bias -> fp32
  proj_gemm<<<dim3(128, 4), 256, 0, stream>>>(ao, wpt, bproj, out);
}
